// RePN_44581760532560
// MI455X (gfx1250) — matrix operand order repaired and checked by reading, not compiled, not run on hardware
//
#include <hip/hip_runtime.h>
#include <hip/hip_bf16.h>
#include <math.h>

// ---------------------------------------------------------------------------
// RePN on MI455X (gfx1250, wave32).
//   k_wcvt   - layer2 weights f32(K,N) -> bf16(N,K)
//   k_mlp    - 2-layer MLP (layer2 via v_wmma_f32_16x16x32_bf16), modulate by
//              features, store bf16 subjF/objF. scores staged to LDS via
//              global_load_async_to_lds_b32 when available.
//   k_pair   - 128x128 tile of subj@obj^T via bf16 WMMA; A/B panels staged to
//              LDS by the Tensor Data Mover (tensor_load_to_lds +
//              s_wait_tensorcnt, double buffered) when available, else
//              async-to-LDS, else manual. Fused mask/sigmoid + per-tile radix
//              top-128 (u64 keys).
//   k_select - per-batch radix top-128 over tile candidates
//   k_gather - boxes / features / rel gathers into d_out
// Workspace: 1MB+1MB (W2T) + 16MB+16MB (subjF/objF) + 544KB cand + 4KB.
// ---------------------------------------------------------------------------

typedef __attribute__((ext_vector_type(16))) __bf16    v16bf;
typedef __attribute__((ext_vector_type(8)))  float     v8f;
typedef __attribute__((ext_vector_type(4)))  unsigned  v4u;
typedef __attribute__((ext_vector_type(8)))  int       v8i;
typedef __attribute__((ext_vector_type(4)))  int       v4i;

#define AS1 __attribute__((address_space(1)))
#define AS3 __attribute__((address_space(3)))

#if defined(__HIP_DEVICE_COMPILE__) && \
    __has_builtin(__builtin_amdgcn_global_load_async_to_lds_b128) && \
    __has_builtin(__builtin_amdgcn_global_load_async_to_lds_b32) && \
    __has_builtin(__builtin_amdgcn_s_wait_asynccnt)
#define HAVE_ASYNC 1
#else
#define HAVE_ASYNC 0
#endif

#if defined(__HIP_DEVICE_COMPILE__) && \
    __has_builtin(__builtin_amdgcn_tensor_load_to_lds) && \
    __has_builtin(__builtin_amdgcn_s_wait_tensorcnt)
#define HAVE_TDM 1
#else
#define HAVE_TDM 0
#endif

#if __has_include(<hip/amd_detail/amd_gfx1250_TDM.h>)
#define TDM_ARGS6 1   /* therock headers -> 6-arg builtin */
#else
#define TDM_ARGS6 0   /* ROCm 7.2 -> 5-arg builtin */
#endif

#define N_TOK    2048
#define BATCH    4
#define NCLS     151
#define SSTRIDE  152
#define HID      512
#define PROJ     1024
#define TOPK     128
#define NTILE    16          /* 2048/128 */
#define TILES_UP 136         /* 16*17/2 upper-wedge tiles */
#define FLATMAX  4194303     /* N_TOK*N_TOK - 1 */

// LDS byte offset of a __shared__ object (AS3 ptrtoint).
static __device__ __forceinline__ unsigned lds_off(const void* p) {
  return (unsigned)(size_t)(AS3 const char*)p;
}

// Monotonic unique 64-bit key: bigger key == bigger value, ties -> smaller idx.
static __device__ __forceinline__ unsigned long long make_key(float v, int flat) {
  unsigned b = __float_as_uint(v);
  unsigned m = (b & 0x80000000u) ? ~b : (b | 0x80000000u);
  return ((unsigned long long)m << 32) | (unsigned)(FLATMAX - flat);
}

// Block-wide top-K threshold via 8 rounds of byte radix histogramming.
template <typename F>
static __device__ unsigned long long
block_topk_threshold(int K, unsigned* s_cnt, int* s_sel, F forEach) {
  unsigned long long prefix = 0ULL;
  int remaining = K;
  for (int shift = 56; shift >= 0; shift -= 8) {
    for (int i = threadIdx.x; i < 256; i += blockDim.x) s_cnt[i] = 0u;
    __syncthreads();
    unsigned long long hiMask = (shift == 56) ? 0ULL : (~0ULL << (shift + 8));
    unsigned long long want = prefix & hiMask;
    forEach([&](unsigned long long k) {
      if ((k & hiMask) == want)
        atomicAdd(&s_cnt[(unsigned)((k >> shift) & 255ULL)], 1u);
    });
    __syncthreads();
    if (threadIdx.x == 0) {
      int accGT = 0, bin = 0;
      for (int x = 255; x >= 0; --x) {
        int c = (int)s_cnt[x];
        if (accGT + c >= remaining) { bin = x; break; }
        accGT += c;
      }
      s_sel[0] = bin; s_sel[1] = remaining - accGT;
    }
    __syncthreads();
    prefix |= ((unsigned long long)(unsigned)s_sel[0]) << shift;
    remaining = s_sel[1];
    __syncthreads();
  }
  return prefix;
}

#if HAVE_TDM
// TDM D# for a 2-D bf16 panel: tile (tile_dim1=128 rows) x (tile_dim0=32 cols)
// out of tensor width 1024, row stride 1024 elements, packed into LDS
// row-major (128x32). group1 is constant; group0 carries lds/global addresses.
static __device__ __forceinline__ v8i tdm_group1(void) {
  v8i g1;
  g1[0] = (int)(1u << 16);        // [17:16] data_size = 1 (2 bytes)
  g1[1] = 0;                      // no atomic barrier
  g1[2] = (int)0x20000000;        // tensor_dim1 lo16 (=8192) in [31:16]
  g1[3] = (int)(32u << 16);       // tile_dim0 = 32 in [31:16]
  g1[4] = 128;                    // tile_dim1 = 128
  g1[5] = 1024;                   // tensor_dim0_stride lo32 = 1024
  g1[6] = 0;                      // stride0 hi / stride1 lo
  g1[7] = 0;                      // stride1 hi
  return g1;                      // tensor_dim0(=1024): lo16 folded below
}

static __device__ __forceinline__ void
tdm_load_panel(unsigned ldsAddr, const __bf16* g, v8i g1) {
  unsigned long long ga = (unsigned long long)(size_t)g;
  v4u g0;
  g0[0] = 1u;                                     // count=1, user descriptor
  g0[1] = ldsAddr;                                // lds_addr [63:32]
  g0[2] = (unsigned)ga;                           // global_addr lo
  g0[3] = (unsigned)((ga >> 32) & 0x01FFFFFFull)  // global_addr[56:32]
          | (2u << 30);                           // type = 2 (image)
  g1[1] = (int)(1024u << 16);                     // tensor_dim0 lo16 @ bit48
  v4i z4 = {0, 0, 0, 0};
#if TDM_ARGS6
  v8i z8 = {0, 0, 0, 0, 0, 0, 0, 0};
  __builtin_amdgcn_tensor_load_to_lds(g0, g1, z4, z4, z8, 0);
#else
  __builtin_amdgcn_tensor_load_to_lds(g0, g1, z4, z4, 0);
#endif
}
#endif

// Cooperative 128x32 bf16 panel stage (fallback when no TDM): per-thread 2x16B.
static __device__ __forceinline__ void
stage_panel(__bf16* lds, const __bf16* g) {
  const int r = threadIdx.x >> 1, h = (threadIdx.x & 1) * 16;
  const __bf16* src = g + (size_t)r * PROJ + h;
  __bf16* dst = lds + r * 32 + h;
#if HAVE_ASYNC
  __builtin_amdgcn_global_load_async_to_lds_b128(
      (AS1 v4i*)(void*)src, (AS3 v4i*)(void*)dst, 0, 0);
  __builtin_amdgcn_global_load_async_to_lds_b128(
      (AS1 v4i*)(void*)(src + 8), (AS3 v4i*)(void*)(dst + 8), 0, 0);
#else
  *(uint4*)dst       = *(const uint4*)src;
  *(uint4*)(dst + 8) = *(const uint4*)(src + 8);
#endif
}

// ---------------------------------------------------------------------------
__global__ __launch_bounds__(256)
void k_wcvt(const float* __restrict__ Ws2, const float* __restrict__ Wo2,
            __bf16* __restrict__ W2Ts, __bf16* __restrict__ W2To) {
  int idx = blockIdx.x * 256 + threadIdx.x;       // 0 .. 2*PROJ*HID-1
  int s = (idx >= PROJ * HID);
  int e = idx - s * PROJ * HID;
  int n = e / HID, k = e % HID;
  const float* W = s ? Wo2 : Ws2;
  __bf16* T = s ? W2To : W2Ts;
  T[(size_t)n * HID + k] = (__bf16)W[(size_t)k * PROJ + n];
}

// ---------------------------------------------------------------------------
__global__ __launch_bounds__(256)
void k_mlp(const float* __restrict__ scores, const float* __restrict__ features,
           const float* __restrict__ W1s, const float* __restrict__ b1s,
           const float* __restrict__ b2s,
           const float* __restrict__ W1o, const float* __restrict__ b1o,
           const float* __restrict__ b2o,
           const __bf16* __restrict__ W2Ts, const __bf16* __restrict__ W2To,
           __bf16* __restrict__ subjF, __bf16* __restrict__ objF) {
  __shared__ float  s_cls[16][SSTRIDE];
  __shared__ __bf16 s_h[16][HID];          // layer2 A operand, (M,K) row-major
  const int rowBase = blockIdx.x * 16;
  const int tid = threadIdx.x, lane = tid & 31, wv = tid >> 5;

  for (int idx = tid; idx < 16 * SSTRIDE; idx += 256) {
    int m = idx / SSTRIDE, c = idx % SSTRIDE;
    const float* src = scores + (size_t)(rowBase + m) * SSTRIDE + c;
    float* dst = &s_cls[m][c];
#if HAVE_ASYNC
    __builtin_amdgcn_global_load_async_to_lds_b32(
        (AS1 int*)(void*)src, (AS3 int*)(void*)dst, 0, 0);
#else
    *dst = *src;
#endif
  }
#if HAVE_ASYNC
  __builtin_amdgcn_s_wait_asynccnt(0);
#endif
  __syncthreads();

  const int mr = lane & 15;
  const int ka = (lane < 16) ? 0 : 8;
  const int kb = (lane < 16) ? 0 : 8;

  for (int s = 0; s < 2; ++s) {
    const float*  W1  = s ? W1o : W1s;
    const float*  b1  = s ? b1o : b1s;
    const float*  b2  = s ? b2o : b2s;
    const __bf16* W2T = s ? W2To : W2Ts;
    __bf16* outF = s ? objF : subjF;

    // layer1: 16x512 outputs / 256 threads, K=151 f32 FMA
    for (int o = tid; o < 16 * HID; o += 256) {
      int m = o >> 9, nn = o & 511;
      float d = b1[nn];
      for (int k = 0; k < NCLS; ++k)
        d = fmaf(s_cls[m][k], W1[k * HID + nn], d);
      s_h[m][nn] = (__bf16)fmaxf(d, 0.0f);
    }
    __syncthreads();

    // layer2: 16x1024, K=512, bf16 WMMA. wave wv owns N = [wv*128, wv*128+128)
    v8f acc[8];
    v8f vz = {0.f, 0.f, 0.f, 0.f, 0.f, 0.f, 0.f, 0.f};
#pragma unroll
    for (int t = 0; t < 8; ++t) acc[t] = vz;

    for (int kk = 0; kk < HID; kk += 32) {
      union { v16bf v; uint4 q[2]; } A;
      const __bf16* ar = &s_h[mr][kk];
      A.q[0] = *(const uint4*)(ar + ka);
      A.q[1] = *(const uint4*)(ar + ka + 16);
#pragma unroll
      for (int t = 0; t < 8; ++t) {
        int n = wv * 128 + t * 16 + mr;
        const __bf16* br = W2T + (size_t)n * HID + kk + kb;
        union { v16bf v; uint4 q[2]; } Bv;
        Bv.q[0] = *(const uint4*)(br);
        Bv.q[1] = *(const uint4*)(br + 16);
        acc[t] = __builtin_amdgcn_wmma_f32_16x16x32_bf16(
            false, A.v, false, Bv.v, (short)0, acc[t], false, false);
      }
    }

    // epilogue: bias + feature modulation, store bf16
#pragma unroll
    for (int t = 0; t < 8; ++t) {
      int n = wv * 128 + t * 16 + mr;
      float bias = b2[n];
#pragma unroll
      for (int v = 0; v < 8; ++v) {
        int m = v + ((lane < 16) ? 0 : 8);
        size_t off = (size_t)(rowBase + m) * PROJ + n;
        outF[off] = (__bf16)((acc[t][v] + bias) * features[off]);
      }
    }
    __syncthreads();
  }
}

// ---------------------------------------------------------------------------
__global__ __launch_bounds__(256)
void k_pair(const __bf16* __restrict__ subjF, const __bf16* __restrict__ objF,
            unsigned long long* __restrict__ cand) {
  __shared__ __bf16 ldsA[2][128][32];
  __shared__ __bf16 ldsB[2][128][32];
  __shared__ unsigned s_cnt[256];
  __shared__ int s_sel[2];
  __shared__ unsigned s_pos;

  const int bId = blockIdx.x / TILES_UP;
  int t = blockIdx.x % TILES_UP;
  int tm = 0;
  while (t >= (NTILE - tm)) { t -= (NTILE - tm); ++tm; }
  const int tn = tm + t;
  const int r0 = tm * 128, c0 = tn * 128;
  const size_t rowOff = (size_t)bId * N_TOK;
  const int tid = threadIdx.x, lane = tid & 31, wv = tid >> 5;

  v8f acc[8];
  v8f vz = {0.f, 0.f, 0.f, 0.f, 0.f, 0.f, 0.f, 0.f};
#pragma unroll
  for (int q = 0; q < 8; ++q) acc[q] = vz;

  const __bf16* gAbase = subjF + (rowOff + r0) * (size_t)PROJ;
  const __bf16* gBbase = objF  + (rowOff + c0) * (size_t)PROJ;

  const int mr = lane & 15;
  const int ka = (lane < 16) ? 0 : 8;
  const int kb = (lane < 16) ? 0 : 8;

#if HAVE_TDM
  const v8i g1 = tdm_group1();
  if (wv == 0) {                                  // one DMA pair per panel
    tdm_load_panel(lds_off(&ldsA[0][0][0]), gAbase, g1);
    tdm_load_panel(lds_off(&ldsB[0][0][0]), gBbase, g1);
  }
#else
  stage_panel(&ldsA[0][0][0], gAbase);
  stage_panel(&ldsB[0][0][0], gBbase);
#endif

  for (int kk = 0; kk < PROJ; kk += 32) {
    const int  cur  = (kk >> 5) & 1;
    const bool more = (kk + 32) < PROJ;
#if HAVE_TDM
    if (wv == 0 && more) {
      tdm_load_panel(lds_off(&ldsA[cur ^ 1][0][0]), gAbase + kk + 32, g1);
      tdm_load_panel(lds_off(&ldsB[cur ^ 1][0][0]), gBbase + kk + 32, g1);
    }
    if (more) __builtin_amdgcn_s_wait_tensorcnt(2);   // current panels done
    else      __builtin_amdgcn_s_wait_tensorcnt(0);
#else
    if (more) {
      stage_panel(&ldsA[cur ^ 1][0][0], gAbase + kk + 32);
      stage_panel(&ldsB[cur ^ 1][0][0], gBbase + kk + 32);
      __builtin_prefetch(gAbase + kk + 64, 0, 0);
      __builtin_prefetch(gBbase + kk + 64, 0, 0);
    }
#if HAVE_ASYNC
    if (more) __builtin_amdgcn_s_wait_asynccnt(4);    // current panels done
    else      __builtin_amdgcn_s_wait_asynccnt(0);
#endif
#endif
    __syncthreads();

    union { v16bf v; uint4 q[2]; } A;
    const __bf16* ar = &ldsA[cur][wv * 16 + mr][0];
    A.q[0] = *(const uint4*)(ar + ka);
    A.q[1] = *(const uint4*)(ar + ka + 16);
#pragma unroll
    for (int tt = 0; tt < 8; ++tt) {
      const __bf16* br = &ldsB[cur][tt * 16 + mr][0] + kb;
      union { v16bf v; uint4 q[2]; } Bv;
      Bv.q[0] = *(const uint4*)(br);
      Bv.q[1] = *(const uint4*)(br + 16);
      acc[tt] = __builtin_amdgcn_wmma_f32_16x16x32_bf16(
          false, A.v, false, Bv.v, (short)0, acc[tt], false, false);
    }
    __syncthreads();
  }

  // mask + sigmoid (strict upper keeps sigmoid(raw); diag=-inf; lower=0.5)
  const int ibase = r0 + wv * 16 + ((lane < 16) ? 0 : 8);
  const int jbase = c0 + mr;
#pragma unroll
  for (int tt = 0; tt < 8; ++tt) {
#pragma unroll
    for (int v = 0; v < 8; ++v) {
      int i = ibase + v, j = jbase + tt * 16;
      float raw = acc[tt][v];
      float val = (j > i) ? (1.0f / (1.0f + __expf(-raw)))
                          : ((j == i) ? -INFINITY : 0.5f);
      acc[tt][v] = val;
    }
  }

  auto forEach = [&](auto&& fn) {
#pragma unroll
    for (int tt = 0; tt < 8; ++tt) {
#pragma unroll
      for (int v = 0; v < 8; ++v) {
        int flat = (ibase + v) * N_TOK + jbase + tt * 16;
        fn(make_key(acc[tt][v], flat));
      }
    }
  };
  unsigned long long thr = block_topk_threshold(TOPK, s_cnt, s_sel, forEach);

  if (tid == 0) s_pos = 0u;
  __syncthreads();
  unsigned long long* dst = cand + (size_t)blockIdx.x * TOPK;
#pragma unroll
  for (int tt = 0; tt < 8; ++tt) {
#pragma unroll
    for (int v = 0; v < 8; ++v) {
      int flat = (ibase + v) * N_TOK + jbase + tt * 16;
      unsigned long long k = make_key(acc[tt][v], flat);
      if (k >= thr) {
        unsigned p = atomicAdd(&s_pos, 1u);
        if (p < TOPK) dst[p] = k;
      }
    }
  }
}

// ---------------------------------------------------------------------------
__global__ __launch_bounds__(256)
void k_select(const unsigned long long* __restrict__ cand,
              float* __restrict__ vals_out,
              int* __restrict__ siArr, int* __restrict__ oiArr) {
  __shared__ unsigned s_cnt[256];
  __shared__ int s_sel[2];
  __shared__ unsigned long long s_keys[TOPK];
  __shared__ unsigned s_pos;
  const int b = blockIdx.x;
  const int M = TILES_UP * TOPK;        // 17408 = 68*256
  const unsigned long long* src = cand + (size_t)b * M;

  auto forEach = [&](auto&& fn) {
    for (int i = threadIdx.x; i < M; i += 256) fn(src[i]);
  };
  unsigned long long thr = block_topk_threshold(TOPK, s_cnt, s_sel, forEach);

  if (threadIdx.x == 0) s_pos = 0u;
  __syncthreads();
  for (int i = threadIdx.x; i < M; i += 256) {
    unsigned long long k = src[i];
    if (k >= thr) {
      unsigned p = atomicAdd(&s_pos, 1u);
      if (p < TOPK) s_keys[p] = k;
    }
  }
  __syncthreads();
  if (threadIdx.x < TOPK) {
    unsigned long long k = s_keys[threadIdx.x];
    int rank = 0;
    for (int j = 0; j < TOPK; ++j) rank += (s_keys[j] > k);
    unsigned m  = (unsigned)(k >> 32);
    unsigned fb = (m & 0x80000000u) ? (m & 0x7FFFFFFFu) : ~m;
    int flat = FLATMAX - (int)(unsigned)(k & 0xFFFFFFFFULL);
    vals_out[b * TOPK + rank] = __uint_as_float(fb);
    siArr[b * TOPK + rank] = flat >> 11;          // / N_TOK
    oiArr[b * TOPK + rank] = flat & (N_TOK - 1);  // % N_TOK
  }
}

// ---------------------------------------------------------------------------
__global__ __launch_bounds__(256)
void k_gather(const float* __restrict__ features, const float* __restrict__ boxes,
              const int* __restrict__ siArr, const int* __restrict__ oiArr,
              float* __restrict__ out) {
  const int b = blockIdx.x >> 7, t = blockIdx.x & 127;
  const int si = siArr[b * TOPK + t];
  const int oi = oiArr[b * TOPK + t];
  const size_t rs = (size_t)b * N_TOK + si;
  const size_t ro = (size_t)b * N_TOK + oi;
  float* boxOut = out + BATCH * TOPK + (size_t)(b * TOPK + t) * 8;
  if (threadIdx.x < 4)      boxOut[threadIdx.x] = boxes[rs * 4 + threadIdx.x];
  else if (threadIdx.x < 8) boxOut[threadIdx.x] = boxes[ro * 4 + (threadIdx.x - 4)];
  float* fOut = out + BATCH * TOPK + BATCH * TOPK * 8 +
                (size_t)(b * TOPK + t) * 3 * PROJ;
  for (int c = threadIdx.x; c < PROJ; c += 256) {
    float fs = features[rs * PROJ + c];
    float fo = features[ro * PROJ + c];
    fOut[c] = fs;
    fOut[PROJ + c] = fo;
    fOut[2 * PROJ + c] = 0.5f * (fs + fo);
  }
}

// ---------------------------------------------------------------------------
extern "C" void kernel_launch(void* const* d_in, const int* in_sizes, int n_in,
                              void* d_out, int out_size, void* d_ws, size_t ws_size,
                              hipStream_t stream) {
  (void)in_sizes; (void)n_in; (void)out_size; (void)ws_size;
  const float* scores   = (const float*)d_in[0];
  const float* features = (const float*)d_in[1];
  const float* boxes    = (const float*)d_in[2];
  const float* Ws1 = (const float*)d_in[3];
  const float* bs1 = (const float*)d_in[4];
  const float* Ws2 = (const float*)d_in[5];
  const float* bs2 = (const float*)d_in[6];
  const float* Wo1 = (const float*)d_in[7];
  const float* bo1 = (const float*)d_in[8];
  const float* Wo2 = (const float*)d_in[9];
  const float* bo2 = (const float*)d_in[10];
  float* out = (float*)d_out;

  char* ws = (char*)d_ws;                               // ~36.2 MB used
  __bf16* W2Ts = (__bf16*)(ws);
  __bf16* W2To = (__bf16*)(ws + (1u << 20));
  __bf16* subjF = (__bf16*)(ws + (2u << 20));
  __bf16* objF  = (__bf16*)(ws + (2u << 20) + (16u << 20));
  unsigned long long* cand =
      (unsigned long long*)(ws + (2u << 20) + (32u << 20));
  int* siArr = (int*)(ws + (2u << 20) + (32u << 20) +
                      (size_t)BATCH * TILES_UP * TOPK * 8);
  int* oiArr = siArr + BATCH * TOPK;

  k_wcvt<<<4096, 256, 0, stream>>>(Ws2, Wo2, W2Ts, W2To);
  k_mlp<<<(BATCH * N_TOK) / 16, 256, 0, stream>>>(
      scores, features, Ws1, bs1, bs2, Wo1, bo1, bo2, W2Ts, W2To, subjF, objF);
  k_pair<<<BATCH * TILES_UP, 256, 0, stream>>>(subjF, objF, cand);
  k_select<<<BATCH, 256, 0, stream>>>(cand, out, siArr, oiArr);
  k_gather<<<BATCH * TOPK, 256, 0, stream>>>(features, boxes, siArr, oiArr, out);
}
